// MeantimeMultiHeadAttention_34780645163123
// MI455X (gfx1250) — compile-verified, matrix operand order checked by reading, not run on hardware
//
#include <hip/hip_runtime.h>
#include <hip/hip_bf16.h>

// ---------------- problem constants ----------------
constexpr int Bc = 8, Sc = 256, Hc = 128, Lc = 4, HDc = 32, LAc = 2;
constexpr int BSc = Bc * Sc;                       // 2048
constexpr float SCALEc = 0.17677669529663687f;     // 1/sqrt(32)

// ---------------- workspace layout (floats) ----------------
constexpr size_t OFF_Q     = 0;                                     // [B,S,H]
constexpr size_t OFF_K     = (size_t)BSc * Hc;                      // [B,S,H]
constexpr size_t OFF_V     = OFF_K * 2;                             // [B,S,H]
constexpr size_t OFF_AK    = (size_t)BSc * Hc * 3;                  // [2,B,S,HD]
constexpr size_t OFF_AV    = OFF_AK + (size_t)2 * BSc * HDc;        // [2,B,S,HD]
constexpr size_t OFF_WT    = OFF_AV + (size_t)2 * BSc * HDc;        // [2,B,S,H]
constexpr size_t OFF_PROBS = OFF_WT + (size_t)2 * BSc * Hc;         // [B,L,S,S]
constexpr size_t OFF_T     = OFF_PROBS + (size_t)Bc * Lc * Sc * Sc; // [2,B,S,H]
constexpr size_t OFF_CTX   = OFF_T + (size_t)2 * BSc * Hc;          // [B,S,H]

// LDS layout for the fused rel kernel
constexpr int RSTR = 132;  // padded rel row stride (floats): 528B, 16B-aligned, 4-bank skew
constexpr size_t REL_SH_FLOATS =
    (size_t)Sc * RSTR + Sc + Hc + HDc + 16 + 128;   // relS, pS, wtS, qS, redS, tpart
constexpr size_t REL_SH_BYTES = REL_SH_FLOATS * sizeof(float);  // ~137.5 KB

// ---------------- WMMA fragment helpers (wave32, 16x16x32 f16) ----------------
typedef __attribute__((ext_vector_type(16))) _Float16 v16h;
typedef __attribute__((ext_vector_type(8)))  float    v8f;

// A[m,k] 16x32 fragment from row-major fp32 (lda in floats).
__device__ __forceinline__ v16h load_a_f32(const float* A, int lda, int row0, int k0) {
  const int lane = threadIdx.x & 31;
  const int m = lane & 15;
  const int half = lane >> 4;
  const float* p = A + (size_t)(row0 + m) * lda + k0 + half * 8;
  v16h a;
#pragma unroll
  for (int e = 0; e < 16; ++e) {
    const int k = (e & 7) + ((e >> 3) << 4);
    a[e] = (_Float16)p[k];
  }
  return a;
}

// B[k,n] 32x16 fragment from row-major fp32 W[K,N] (ldw in floats).
__device__ __forceinline__ v16h load_b_f32(const float* W, int ldw, int k0, int col0) {
  const int lane = threadIdx.x & 31;
  const int n = lane & 15;
  const int half = lane >> 4;
  const float* p = W + (size_t)(k0 + half * 8) * ldw + col0 + n;
  v16h b;
#pragma unroll
  for (int e = 0; e < 16; ++e) {
    const int k = (e & 7) + ((e >> 3) << 4);
    b[e] = (_Float16)p[(size_t)k * ldw];
  }
  return b;
}

// B[k,n] fragment where source is stored transposed: M[n, k] row-major.
__device__ __forceinline__ v16h load_bt_f32(const float* Mt, int ldk, int k0, int col0) {
  const int lane = threadIdx.x & 31;
  const int n = lane & 15;
  const int half = lane >> 4;
  const float* p = Mt + (size_t)(col0 + n) * ldk + k0 + half * 8;
  v16h b;
#pragma unroll
  for (int e = 0; e < 16; ++e) {
    const int k = (e & 7) + ((e >> 3) << 4);
    b[e] = (_Float16)p[k];
  }
  return b;
}

__device__ __forceinline__ v8f wmma16(v16h a, v16h b, v8f c) {
  return __builtin_amdgcn_wmma_f32_16x16x32_f16(false, a, false, b, (short)0, c,
                                                false, false);
}

__device__ __forceinline__ float wave_max(float v) {
  v = fmaxf(v, __shfl_xor(v, 16));
  v = fmaxf(v, __shfl_xor(v, 8));
  v = fmaxf(v, __shfl_xor(v, 4));
  v = fmaxf(v, __shfl_xor(v, 2));
  v = fmaxf(v, __shfl_xor(v, 1));
  return v;
}
__device__ __forceinline__ float wave_sum(float v) {
  v += __shfl_xor(v, 16);
  v += __shfl_xor(v, 8);
  v += __shfl_xor(v, 4);
  v += __shfl_xor(v, 2);
  v += __shfl_xor(v, 1);
  return v;
}

// ---------------- K1: fused Q/K/V projection ----------------
__global__ void __launch_bounds__(32) k_qkv(
    const float* __restrict__ hs,
    const float* __restrict__ Wq, const float* __restrict__ bq,
    const float* __restrict__ Wk, const float* __restrict__ bk,
    const float* __restrict__ Wv, const float* __restrict__ bv,
    float* __restrict__ ws) {
  const int row0 = blockIdx.x * 16;
  const int mat  = blockIdx.y >> 3;
  const int n0   = (blockIdx.y & 7) * 16;
  const float* W    = (mat == 0) ? Wq : (mat == 1) ? Wk : Wv;
  const float* bias = (mat == 0) ? bq : (mat == 1) ? bk : bv;
  float* dst = ws + (size_t)mat * BSc * Hc;
  v8f acc = {};
#pragma unroll
  for (int kk = 0; kk < Hc; kk += 32) {
    v16h a = load_a_f32(hs, Hc, row0, kk);
    v16h b = load_b_f32(W, Hc, kk, n0);
    acc = wmma16(a, b, acc);
  }
  const int lane = threadIdx.x & 31;
  const int n = lane & 15, half = lane >> 4;
  const float bb = bias[n0 + n];
#pragma unroll
  for (int v = 0; v < 8; ++v)
    dst[(size_t)(row0 + v + half * 8) * Hc + n0 + n] = acc[v] + bb;
}

// ---------------- K2: absolute-kernel K/V projections ----------------
__global__ void __launch_bounds__(32) k_abskv(
    const float* __restrict__ abs0, const float* __restrict__ abs1,
    const float* __restrict__ Wk, const float* __restrict__ bk,
    const float* __restrict__ Wv, const float* __restrict__ bv,
    float* __restrict__ ws) {
  const int row0 = blockIdx.x * 16;
  const int kv = blockIdx.y >> 2;
  const int i  = (blockIdx.y >> 1) & 1;
  const int n0 = (blockIdx.y & 1) * 16;
  const float* A    = i ? abs1 : abs0;
  const float* W    = kv ? Wv : Wk;
  const float* bias = kv ? bv : bk;
  float* dst = ws + (kv ? OFF_AV : OFF_AK) + (size_t)i * BSc * HDc;
  const int col0 = i * HDc + n0;
  v8f acc = {};
#pragma unroll
  for (int kk = 0; kk < Hc; kk += 32) {
    v16h a = load_a_f32(A, Hc, row0, kk);
    v16h b = load_b_f32(W, Hc, kk, col0);
    acc = wmma16(a, b, acc);
  }
  const int lane = threadIdx.x & 31;
  const int n = lane & 15, half = lane >> 4;
  const float bb = bias[col0 + n];
#pragma unroll
  for (int v = 0; v < 8; ++v)
    dst[(size_t)(row0 + v + half * 8) * HDc + n0 + n] = acc[v] + bb;
}

// ---------------- K3: w~[i,r,c] = sum_d Wk[c,(2+i)*HD+d] * Q[r,(2+i)*HD+d] ----------------
__global__ void __launch_bounds__(32) k_wtilde(const float* __restrict__ Wk,
                                               float* __restrict__ ws) {
  const int row0 = blockIdx.x * 16;
  const int i = blockIdx.y >> 3;
  const int c0 = (blockIdx.y & 7) * 16;
  const int hcol = (LAc + i) * HDc;
  const float* Q = ws + OFF_Q;
  v16h a = load_a_f32(Q + hcol, Hc, row0, 0);
  v16h b = load_bt_f32(Wk + hcol, Hc, 0, c0);
  v8f acc = {};
  acc = wmma16(a, b, acc);
  float* wt = ws + OFF_WT + (size_t)i * BSc * Hc;
  const int lane = threadIdx.x & 31;
  const int n = lane & 15, half = lane >> 4;
#pragma unroll
  for (int v = 0; v < 8; ++v)
    wt[(size_t)(row0 + v + half * 8) * Hc + c0 + n] = acc[v];
}

// ---------------- K4: abs-head scores (WMMA) + softmax ----------------
// grid (S/16, 2, B), block 256 (8 waves, two 16x16 score tiles per wave).
__global__ void __launch_bounds__(256) k_abs_scores(
    const float* __restrict__ mask, float* __restrict__ ws) {
  __shared__ float sc[16][Sc];
  __shared__ float red[16][16];
  __shared__ float rowmax[16];
  __shared__ float rowsum[16];

  const int q0 = blockIdx.x * 16;
  const int l  = blockIdx.y;          // 0 or 1 (absolute heads only)
  const int b  = blockIdx.z;
  const int tid = threadIdx.x;
  const int wave = tid >> 5;
  const int lane = tid & 31;

  const float* Q  = ws + OFF_Q + (size_t)b * Sc * Hc + l * HDc;
  const float* K  = ws + OFF_K + (size_t)b * Sc * Hc + l * HDc;
  const float* aK = ws + OFF_AK + (size_t)(l * BSc + b * Sc) * HDc;

  // scores = (Q@K^T + Q@aK^T) * SCALE + mask
  {
    v16h a = load_a_f32(Q, Hc, q0, 0);
#pragma unroll
    for (int t = 0; t < 2; ++t) {
      const int n0 = (wave * 2 + t) * 16;
      v8f acc = {};
      acc = wmma16(a, load_bt_f32(K, Hc, 0, n0), acc);
      acc = wmma16(a, load_bt_f32(aK, HDc, 0, n0), acc);
      const int n = lane & 15, half = lane >> 4;
#pragma unroll
      for (int v = 0; v < 8; ++v) {
        const int m = v + half * 8;
        sc[m][n0 + n] = acc[v] * SCALEc +
                        mask[((size_t)b * Sc + (q0 + m)) * Sc + n0 + n];
      }
    }
  }
  __syncthreads();

  // softmax per row
  const int r  = tid >> 4;
  const int c0 = (tid & 15) * 16;
  float mx = -3.4e38f;
#pragma unroll
  for (int j = 0; j < 16; ++j) mx = fmaxf(mx, sc[r][c0 + j]);
  red[r][tid & 15] = mx;
  __syncthreads();
  if (tid < 16) {
    float m2 = -3.4e38f;
#pragma unroll
    for (int j = 0; j < 16; ++j) m2 = fmaxf(m2, red[tid][j]);
    rowmax[tid] = m2;
  }
  __syncthreads();
  const float rm = rowmax[r];
  float e[16];
  float s = 0.f;
#pragma unroll
  for (int j = 0; j < 16; ++j) { e[j] = __expf(sc[r][c0 + j] - rm); s += e[j]; }
  red[r][tid & 15] = s;
  __syncthreads();
  if (tid < 16) {
    float s2 = 0.f;
#pragma unroll
    for (int j = 0; j < 16; ++j) s2 += red[tid][j];
    rowsum[tid] = s2;
  }
  __syncthreads();
  const float inv = 1.0f / rowsum[r];
  float* probs = ws + OFF_PROBS + (((size_t)b * Lc + l) * Sc + q0 + r) * Sc + c0;
#pragma unroll
  for (int j = 0; j < 16; ++j) probs[j] = e[j] * inv;
}

// ---------------- K5: fused rel-head scores + softmax + t, single rel read ----------------
// grid (S, 2, B), block 256. Stages rel[b,q,:,:] (128 KB) in LDS via the CDNA5
// async global->LDS path (ASYNCcnt-tracked, no VGPR round-trip).
__global__ void __launch_bounds__(256) k_rel_fused(
    const float* __restrict__ mask,
    const float* __restrict__ rel0, const float* __restrict__ rel1,
    float* __restrict__ ws) {
  extern __shared__ float smem[];
  float* relS  = smem;                       // Sc * RSTR
  float* pS    = relS + (size_t)Sc * RSTR;   // Sc
  float* wtS   = pS + Sc;                    // Hc
  float* qS    = wtS + Hc;                   // HDc
  float* redS  = qS + HDc;                   // 16
  float* tpart = redS + 16;                  // 128

  const int q = blockIdx.x;
  const int i = blockIdx.y;
  const int b = blockIdx.z;
  const int l = LAc + i;
  const int tid = threadIdx.x;
  const int lane = tid & 31;
  const int wave = tid >> 5;

  const float* rel = i ? rel1 : rel0;
  const float* rbase = rel + (size_t)(b * Sc + q) * Sc * Hc;

  // Stage rel slice into LDS with padded rows using GLOBAL_LOAD_ASYNC_TO_LDS_B128.
  // VDST = 32-bit LDS byte address (low 32 bits of the generic pointer),
  // VADDR = 64-bit global address; completion tracked by ASYNCcnt.
  for (int f4 = tid; f4 < (Sc * Hc) / 4; f4 += 256) {
    const int k  = f4 >> 5;     // 32 float4 per 128-float row
    const int c4 = f4 & 31;
    const float* src = rbase + (size_t)f4 * 4;
    unsigned lds_addr = (unsigned)(uintptr_t)(void*)(relS + (size_t)k * RSTR + c4 * 4);
    asm volatile("global_load_async_to_lds_b128 %0, %1, off"
                 :
                 : "v"(lds_addr), "v"(src)
                 : "memory");
  }
  if (tid < Hc)  wtS[tid] = ws[OFF_WT + ((size_t)i * BSc + b * Sc + q) * Hc + tid];
  if (tid < HDc) qS[tid]  = ws[OFF_Q + ((size_t)(b * Sc + q)) * Hc + l * HDc + tid];
  asm volatile("s_wait_asynccnt 0" ::: "memory");
  __syncthreads();

  // Thread k: score[k] = (Q.K[k] + rel[k,:].w~) * SCALE + mask
  const int k = tid;
  const float* Krow = ws + OFF_K + ((size_t)(b * Sc + k)) * Hc + l * HDc;
  float qk = 0.f;
#pragma unroll
  for (int d = 0; d < HDc; ++d) qk += qS[d] * Krow[d];
  float rterm = 0.f;
  const float* rrow = relS + (size_t)k * RSTR;
  for (int c = 0; c < Hc; ++c) rterm += rrow[c] * wtS[c];
  const float val = (qk + rterm) * SCALEc + mask[((size_t)b * Sc + q) * Sc + k];

  // Block softmax over 256 elements.
  float m = wave_max(val);
  if (lane == 0) redS[wave] = m;
  __syncthreads();
  if (tid == 0) {
    float mm = redS[0];
#pragma unroll
    for (int w = 1; w < 8; ++w) mm = fmaxf(mm, redS[w]);
    redS[8] = mm;
  }
  __syncthreads();
  float p = __expf(val - redS[8]);
  float s = wave_sum(p);
  if (lane == 0) redS[wave] = s;
  __syncthreads();
  if (tid == 0) {
    float ss = 0.f;
#pragma unroll
    for (int w = 1; w < 8; ++w) ss += redS[w];
    redS[9] = ss + redS[0];
  }
  __syncthreads();
  p *= 1.0f / redS[9];
  pS[k] = p;
  ws[OFF_PROBS + (((size_t)b * Lc + l) * Sc + q) * Sc + k] = p;
  __syncthreads();

  // t[c] = sum_k p[k] * rel[k,c]  (from LDS; the rel tensor is never re-read from HBM)
  const int c  = tid & 127;
  const int kh = tid >> 7;
  float acc = 0.f;
  const int kbeg = kh * 128;
  for (int kk = kbeg; kk < kbeg + 128; ++kk)
    acc += pS[kk] * relS[(size_t)kk * RSTR + c];
  if (kh) tpart[c] = acc;
  __syncthreads();
  if (!kh)
    ws[OFF_T + ((size_t)i * BSc + b * Sc + q) * Hc + c] = acc + tpart[c];
}

// ---------------- K6: ctx = probs@V (+ probs@aV | t@Wv_slice) ----------------
__global__ void __launch_bounds__(64) k_ctx(const float* __restrict__ Wv,
                                            float* __restrict__ ws) {
  const int q0 = blockIdx.x * 16;
  const int l  = blockIdx.y;
  const int b  = blockIdx.z;
  const int wave = threadIdx.x >> 5;
  const int n0 = wave * 16;
  const float* probs = ws + OFF_PROBS + ((size_t)b * Lc + l) * Sc * Sc;
  const float* V = ws + OFF_V + (size_t)b * Sc * Hc + l * HDc;
  v8f acc = {};
  if (l < LAc) {
    const float* aV = ws + OFF_AV + (size_t)(l * BSc + b * Sc) * HDc;
    for (int kk = 0; kk < Sc; kk += 32) {
      v16h a = load_a_f32(probs, Sc, q0, kk);   // shared A for both WMMAs
      acc = wmma16(a, load_b_f32(V, Hc, kk, n0), acc);
      acc = wmma16(a, load_b_f32(aV, HDc, kk, n0), acc);
    }
  } else {
    for (int kk = 0; kk < Sc; kk += 32) {
      v16h a = load_a_f32(probs, Sc, q0, kk);
      acc = wmma16(a, load_b_f32(V, Hc, kk, n0), acc);
    }
    const int i = l - LAc;
    const float* t = ws + OFF_T + (size_t)(i * BSc + b * Sc) * Hc;
    const float* Wvs = Wv + l * HDc;   // B[c,n] = Wv[c*H + l*HD + n]
#pragma unroll
    for (int kk = 0; kk < Hc; kk += 32) {
      v16h a = load_a_f32(t, Hc, q0, kk);
      acc = wmma16(a, load_b_f32(Wvs, Hc, kk, n0), acc);
    }
  }
  const int lane = threadIdx.x & 31;
  const int n = lane & 15, half = lane >> 4;
  float* ctx = ws + OFF_CTX;
#pragma unroll
  for (int v = 0; v < 8; ++v)
    ctx[(size_t)(b * Sc + q0 + v + half * 8) * Hc + l * HDc + n0 + n] = acc[v];
}

// ---------------- K7: out = ctx @ Wo + bo ----------------
__global__ void __launch_bounds__(32) k_out(const float* __restrict__ Wo,
                                            const float* __restrict__ bo,
                                            const float* __restrict__ ws,
                                            float* __restrict__ out) {
  const int row0 = blockIdx.x * 16;
  const int n0 = blockIdx.y * 16;
  const float* ctx = ws + OFF_CTX;
  v8f acc = {};
#pragma unroll
  for (int kk = 0; kk < Hc; kk += 32) {
    v16h a = load_a_f32(ctx, Hc, row0, kk);
    v16h b = load_b_f32(Wo, Hc, kk, n0);
    acc = wmma16(a, b, acc);
  }
  const int lane = threadIdx.x & 31;
  const int n = lane & 15, half = lane >> 4;
  const float bb = bo[n0 + n];
#pragma unroll
  for (int v = 0; v < 8; ++v)
    out[(size_t)(row0 + v + half * 8) * Hc + n0 + n] = acc[v] + bb;
}

// ---------------- launcher ----------------
extern "C" void kernel_launch(void* const* d_in, const int* in_sizes, int n_in,
                              void* d_out, int out_size, void* d_ws, size_t ws_size,
                              hipStream_t stream) {
  const float* hs   = (const float*)d_in[0];
  const float* mask = (const float*)d_in[1];
  const float* abs0 = (const float*)d_in[2];
  const float* abs1 = (const float*)d_in[3];
  const float* rel0 = (const float*)d_in[4];
  const float* rel1 = (const float*)d_in[5];
  const float* Wq = (const float*)d_in[6];
  const float* bq = (const float*)d_in[7];
  const float* Wk = (const float*)d_in[8];
  const float* bk = (const float*)d_in[9];
  const float* Wv = (const float*)d_in[10];
  const float* bv = (const float*)d_in[11];
  const float* Wo = (const float*)d_in[12];
  const float* bo = (const float*)d_in[13];
  float* out = (float*)d_out;
  float* ws  = (float*)d_ws;

  k_qkv<<<dim3(BSc / 16, 24), 32, 0, stream>>>(hs, Wq, bq, Wk, bk, Wv, bv, ws);
  k_abskv<<<dim3(BSc / 16, 8), 32, 0, stream>>>(abs0, abs1, Wk, bk, Wv, bv, ws);
  k_wtilde<<<dim3(BSc / 16, 16), 32, 0, stream>>>(Wk, ws);
  k_abs_scores<<<dim3(Sc / 16, LAc, Bc), 256, 0, stream>>>(mask, ws);
  k_rel_fused<<<dim3(Sc, 2, Bc), 256, REL_SH_BYTES, stream>>>(mask, rel0, rel1, ws);
  k_ctx<<<dim3(Sc / 16, Lc, Bc), 64, 0, stream>>>(Wv, ws);
  k_out<<<dim3(BSc / 16, Hc / 16), 32, 0, stream>>>(Wo, bo, ws, out);
}
